// memory_retrive_sentiment_61529701482983
// MI455X (gfx1250) — matrix operand order repaired and checked by reading, not compiled, not run on hardware
//
#include <hip/hip_runtime.h>

// Problem constants (from reference)
#define L_DIM  50
#define B_DIM  1024
#define D_DIM  512
#define HD_DIM 2048   // H * DK
#define N_HOP  3

// ---------------------------------------------------------------------------
// Math note: softmax over the singleton key axis == 1.0, so attention output
// is exactly v, independent of momery/q/k. temp = (hid@wv+bv)@wo+bo is hop-
// invariant, so the 3-hop loop telescopes to: out = momery + 3*w*t.
// Only two small L2-resident GEMMs + one 210MB stream (≈9us @ 23.3TB/s) live.
// ---------------------------------------------------------------------------

typedef __bf16 v16bf __attribute__((ext_vector_type(16)));
typedef float  v8f   __attribute__((ext_vector_type(8)));

union FragU {
    uint4 q[2];   // two 16B LDS reads (ds_load_b128)
    v16bf v;      // 16 bf16 WMMA fragment
};

__device__ __forceinline__ unsigned short f2bf(float f) {
    // round-to-nearest-even fp32 -> bf16
    unsigned int u = __builtin_bit_cast(unsigned int, f);
    u += 0x7FFFu + ((u >> 16) & 1u);
    return (unsigned short)(u >> 16);
}

__device__ __forceinline__ unsigned int f2bf2(float lo, float hi) {
    return (unsigned int)f2bf(lo) | ((unsigned int)f2bf(hi) << 16);
}

// C[M,N] = A[M,K] @ B[K,N] + bias[N]
// A: bf16 (workspace) or f32 (global, converted on LDS stage); B: f32.
// Block = 256 threads (8 waves), block tile 64(M) x 32(N), wave tile 16x16.
// LDS rows are 40 bf16 (80B: 16B-aligned, odd bank-group stride).
template <bool A_BF16, bool OUT_BF16>
__global__ __launch_bounds__(256)
void gemm_bf16_wmma(const void* __restrict__ Ap, const float* __restrict__ Bp,
                    const float* __restrict__ bias, void* __restrict__ Cp,
                    int M, int N, int K)
{
    __shared__ unsigned int As_u[64][20];   // A tile: [m][k] packed 2xbf16/uint
    __shared__ unsigned int Bs_u[32][20];   // B tile TRANSPOSED: [n][k]

    const int tid  = threadIdx.x;
    const int lane = tid & 31;
    const int wave = tid >> 5;
    const int wm   = wave >> 1;           // 0..3  (M sub-tile)
    const int wn   = wave & 1;            // 0..1  (N sub-tile)
    const int blockM = blockIdx.y * 64;
    const int blockN = blockIdx.x * 32;

    v8f acc = {0.f, 0.f, 0.f, 0.f, 0.f, 0.f, 0.f, 0.f};

    // CDNA5 16x16x32 bf16 fragment coordinates:
    // A: lane l -> row l&15; K-halves (l>>4)*8..+7 and +16..+23
    const int a_row = wm * 16 + (lane & 15);
    const int aq    = (lane >> 4);          // uint4 index within row: 0 or 1
    // B: lane l -> col l&15; K-halves (l>>4)*8..+7 and +16..+23
    const int b_row = wn * 16 + (lane & 15);
    const int bq    = (lane >> 4);          // uint4 index within row: 0 or 1

    // staging coordinates
    const int sa_r  = tid >> 2;             // 0..63  A row
    const int sa_cb = (tid & 3) * 4;        // A uint col base (8 bf16)
    const int sb_r  = tid >> 3;             // 0..31  B k-row
    const int sb_c  = (tid & 7) * 4;        // B col base

    for (int k0 = 0; k0 < K; k0 += 32) {
        // ---- stage A tile (64 rows x 32 k) -> LDS packed bf16 ----
        if (A_BF16) {
            const uint4 d = *(const uint4*)((const unsigned short*)Ap +
                            (size_t)(blockM + sa_r) * K + k0 + sa_cb * 2);
            As_u[sa_r][sa_cb + 0] = d.x;
            As_u[sa_r][sa_cb + 1] = d.y;
            As_u[sa_r][sa_cb + 2] = d.z;
            As_u[sa_r][sa_cb + 3] = d.w;
        } else {
            const float* src = (const float*)Ap + (size_t)(blockM + sa_r) * K
                               + k0 + sa_cb * 2;
            const float4 f0 = ((const float4*)src)[0];
            const float4 f1 = ((const float4*)src)[1];
            As_u[sa_r][sa_cb + 0] = f2bf2(f0.x, f0.y);
            As_u[sa_r][sa_cb + 1] = f2bf2(f0.z, f0.w);
            As_u[sa_r][sa_cb + 2] = f2bf2(f1.x, f1.y);
            As_u[sa_r][sa_cb + 3] = f2bf2(f1.z, f1.w);
        }
        // ---- stage B tile (32 k x 32 n) -> LDS transposed [n][k] bf16 ----
        {
            const float4 f = *(const float4*)(Bp + (size_t)(k0 + sb_r) * N +
                                              blockN + sb_c);
            unsigned short* bs = (unsigned short*)Bs_u;
            bs[(sb_c + 0) * 40 + sb_r] = f2bf(f.x);
            bs[(sb_c + 1) * 40 + sb_r] = f2bf(f.y);
            bs[(sb_c + 2) * 40 + sb_r] = f2bf(f.z);
            bs[(sb_c + 3) * 40 + sb_r] = f2bf(f.w);
        }
        __syncthreads();

        // ---- fragments: 2x ds_load_b128 each ----
        FragU a, b;
        const uint4* arow = (const uint4*)&As_u[a_row][0];
        a.q[0] = arow[aq];          // K = a_kb .. a_kb+7
        a.q[1] = arow[aq + 2];      // K = a_kb+16 .. a_kb+23
        const uint4* brow = (const uint4*)&Bs_u[b_row][0];
        b.q[0] = brow[bq];          // K = b_kb .. b_kb+7
        b.q[1] = brow[bq + 2];      // K = b_kb+16 .. b_kb+23

        acc = __builtin_amdgcn_wmma_f32_16x16x32_bf16(
                  /*neg_a=*/false, a.v, /*neg_b=*/false, b.v,
                  /*c_mod=*/(short)0, acc, /*reuse_a=*/false, /*reuse_b=*/false);
        __syncthreads();
    }

    // ---- epilogue: D layout = lane l col l&15, VGPR i -> row 8*(l>>4)+i ----
    const int out_col  = blockN + wn * 16 + (lane & 15);
    const int row_base = blockM + wm * 16 + (lane >> 4) * 8;
    const float bn = bias[out_col];
    #pragma unroll
    for (int i = 0; i < 8; ++i) {
        const float  val = acc[i] + bn;
        const size_t idx = (size_t)(row_base + i) * N + out_col;
        if (OUT_BF16) ((unsigned short*)Cp)[idx] = f2bf(val);
        else          ((float*)Cp)[idx] = val;
    }
}

// out[l,b,d] = momery[l,b,d] + (HOP*w) * t[b,d]  — the dominant 210MB stream.
__global__ __launch_bounds__(256)
void fused_memory_update(const float* __restrict__ mem,
                         const float* __restrict__ t,
                         const float* __restrict__ w_ptr,
                         float* __restrict__ out)
{
    const size_t n4  = (size_t)L_DIM * B_DIM * D_DIM / 4;
    const size_t bd4 = (size_t)B_DIM * D_DIM / 4;
    const size_t i   = (size_t)blockIdx.x * blockDim.x + threadIdx.x;
    if (i >= n4) return;

    // prefetch a future chunk of the stream (global_prefetch_b8)
    if (i + 16384 < n4)
        __builtin_prefetch(((const float4*)mem) + i + 16384, 0, 0);

    const float  s  = (float)N_HOP * w_ptr[0];
    const float4 m  = ((const float4*)mem)[i];
    const float4 tv = ((const float4*)t)[i % bd4];   // 2MB, L2-resident x50 reuse
    float4 o;
    o.x = m.x + s * tv.x;
    o.y = m.y + s * tv.y;
    o.z = m.z + s * tv.z;
    o.w = m.w + s * tv.w;
    ((float4*)out)[i] = o;
}

extern "C" void kernel_launch(void* const* d_in, const int* in_sizes, int n_in,
                              void* d_out, int out_size, void* d_ws, size_t ws_size,
                              hipStream_t stream)
{
    // setup_inputs() order:
    // 0 momery, 1 hid, 2 text_polarity, 3 attribute_polarity, 4 w,
    // 5 p_w1, 6 p_b1, 7 p_w2, 8 p_b2, 9 wq, 10 bq, 11 wk, 12 bk,
    // 13 wv, 14 bv, 15 wo, 16 bo
    const float* mem = (const float*)d_in[0];
    const float* hid = (const float*)d_in[1];
    const float* w   = (const float*)d_in[4];
    const float* wv  = (const float*)d_in[13];
    const float* bv  = (const float*)d_in[14];
    const float* wo  = (const float*)d_in[15];
    const float* bo  = (const float*)d_in[16];

    // workspace: v_flat bf16 [1024,2048] (4MB) then t f32 [1024,512] (2MB)
    unsigned short* vflat = (unsigned short*)d_ws;
    float* t = (float*)((char*)d_ws + (size_t)B_DIM * HD_DIM * sizeof(unsigned short));

    // gemm1: v_flat = bf16(hid @ wv + bv)   [1024,512]x[512,2048]
    gemm_bf16_wmma<false, true>
        <<<dim3(HD_DIM / 32, B_DIM / 64), 256, 0, stream>>>(
            hid, wv, bv, vflat, B_DIM, HD_DIM, D_DIM);

    // gemm2: t = v_flat @ wo + bo           [1024,2048]x[2048,512]
    gemm_bf16_wmma<true, false>
        <<<dim3(D_DIM / 32, B_DIM / 64), 256, 0, stream>>>(
            vflat, wo, bo, t, B_DIM, D_DIM, HD_DIM);

    // out = momery + 3*w*t  (dominant stream, ~9us at 23.3 TB/s)
    const size_t n4 = (size_t)L_DIM * B_DIM * D_DIM / 4;
    fused_memory_update<<<(unsigned)((n4 + 255) / 256), 256, 0, stream>>>(
        mem, t, w, (float*)d_out);
}
